// MemAttention_8710193676606
// MI455X (gfx1250) — matrix operand order repaired and checked by reading, not compiled, not run on hardware
//
#include <hip/hip_runtime.h>
#include <math.h>

// ---- shapes ----
#define BN   4        // new frames
#define LBN  28       // buffered frames
#define MN   32       // BN + LBN
#define DK   256
#define DV   32
#define HS   64
#define WS   64
#define PIX  (HS*WS)  // 4096

typedef _Float16 v8h  __attribute__((ext_vector_type(8)));
typedef _Float16 v16h __attribute__((ext_vector_type(16)));
typedef float    v8f  __attribute__((ext_vector_type(8)));

#define SB_PITCH 264   // LDS row pitch in halves: 528B = 33*16B -> conflict-free b128 lanes

// ------------------------------------------------------------------
// 1) Transpose weights Q,K (256x256 f32, [d][e]) -> f16 [e][d]
// ------------------------------------------------------------------
__global__ void k_transpose_w(const float* __restrict__ Q,
                              const float* __restrict__ K,
                              _Float16* __restrict__ Qt,
                              _Float16* __restrict__ Kt) {
    int idx = blockIdx.x * blockDim.x + threadIdx.x;   // idx = e*256 + d
    if (idx >= DK * DK) return;
    int d = idx & (DK - 1);
    int e = idx >> 8;
    Qt[idx] = (_Float16)Q[d * DK + e];
    Kt[idx] = (_Float16)K[d * DK + e];
}

// ------------------------------------------------------------------
// 2) Transpose fc [b][d][p] f32 -> fcT [b][p][d] f16 (LDS 32x32 tiles)
// ------------------------------------------------------------------
__global__ void k_transpose_fc(const float* __restrict__ fc,
                               _Float16* __restrict__ fcT) {
    __shared__ float tile[32][33];
    int b  = blockIdx.z;
    int d0 = blockIdx.y * 32;
    int p0 = blockIdx.x * 32;
    int tx = threadIdx.x;          // 0..31
    int ty = threadIdx.y;          // 0..7
    const float* src = fc + ((size_t)b * DK + d0) * PIX + p0;
#pragma unroll
    for (int i = 0; i < 32; i += 8)
        tile[ty + i][tx] = src[(size_t)(ty + i) * PIX + tx];   // coalesced over p
    __syncthreads();
    _Float16* dst = fcT + ((size_t)b * PIX + p0) * DK + d0;
#pragma unroll
    for (int i = 0; i < 32; i += 8)
        dst[(size_t)(ty + i) * DK + tx] = (_Float16)tile[tx][ty + i]; // coalesced over d
}

// ------------------------------------------------------------------
// 3) WMMA projection:  q[b,e,p] = sum_d Qt[e,d]*fcT[b,p,d]
//                     kp[b,e,p] = sum_d Kt[e,d]*fcT[b,p,d]
// Block = 256 thr = 8 waves, covers 32 pixels (2 p-tiles) x all 256 e.
// fc tile (32 pixels x 256 d f16 = 16KB) staged once per block via
// GLOBAL_LOAD_ASYNC_TO_LDS_B128, then each wave runs 2e x 2p x 2w = 8
// WMMAs per 32-wide K chunk (64 WMMA / wave).
// ------------------------------------------------------------------
__global__ __launch_bounds__(256)
void k_proj_qk(const _Float16* __restrict__ fcT,
               const _Float16* __restrict__ Qt,
               const _Float16* __restrict__ Kt,
               float* __restrict__ q,
               float* __restrict__ kp) {
    __shared__ _Float16 sB[32 * SB_PITCH];   // 32 pixel rows x 256 d (padded)

    const int tid  = threadIdx.x;
    const int lane = tid & 31;
    const int wave = tid >> 5;               // 0..7
    const int b    = blockIdx.x >> 7;        // 0..3
    const int p0   = (blockIdx.x & 127) * 32;

    // ---- async stage: 32 fcT rows (512B each) -> LDS via ASYNCcnt path ----
    {
        const char* gbase = (const char*)(fcT + ((size_t)b * PIX + p0) * DK);
        unsigned ldsbase = (unsigned)(size_t)(&sB[0]);
#pragma unroll
        for (int it = 0; it < 4; ++it) {
            int i   = tid + it * 256;        // 1024 x 16B chunks = 16KB
            int row = i >> 5;                // pixel row 0..31
            int chk = i & 31;                // 16B chunk within row
            unsigned    lds = ldsbase + (unsigned)(row * (SB_PITCH * 2) + chk * 16);
            const char* g   = gbase + (size_t)row * (DK * 2) + chk * 16;
            asm volatile("global_load_async_to_lds_b128 %0, %1, off"
                         :: "v"(lds), "v"(g) : "memory");
        }
        asm volatile("s_wait_asynccnt 0x0" ::: "memory");
        __syncthreads();
    }

    const int e0a = wave * 16;               // first e-tile
    const int e0b = (wave + 8) * 16;         // second e-tile
    const int mrow   = lane & 15;
    const int koff8  = (lane < 16) ? 0 : 8;  // A-operand lane K offset
    const int koff16 = (lane < 16) ? 0 : 16; // unused; B loads use koff8 as well

    const _Float16* arQa = Qt + (size_t)(e0a + mrow) * DK;
    const _Float16* arQb = Qt + (size_t)(e0b + mrow) * DK;
    const _Float16* arKa = Kt + (size_t)(e0a + mrow) * DK;
    const _Float16* arKb = Kt + (size_t)(e0b + mrow) * DK;
    const _Float16* sb0  = &sB[(lane & 15) * SB_PITCH];        // p-tile 0 col
    const _Float16* sb1  = &sB[((lane & 15) + 16) * SB_PITCH]; // p-tile 1 col

    v8f cQa0 = {}, cQa1 = {}, cQb0 = {}, cQb1 = {};
    v8f cKa0 = {}, cKa1 = {}, cKb0 = {}, cKb1 = {};

#pragma unroll
    for (int kc = 0; kc < DK / 32; ++kc) {
        const int d0 = kc * 32;
        const int da = d0 + koff8;
        v8h lo, hi;
        // A fragments (Qt/Kt rows, L2-resident): lane K = {o..o+7, o+16..o+23}
        lo = *(const v8h*)(arQa + da); hi = *(const v8h*)(arQa + da + 16);
        v16h aQa = __builtin_shufflevector(lo, hi, 0,1,2,3,4,5,6,7,8,9,10,11,12,13,14,15);
        lo = *(const v8h*)(arQb + da); hi = *(const v8h*)(arQb + da + 16);
        v16h aQb = __builtin_shufflevector(lo, hi, 0,1,2,3,4,5,6,7,8,9,10,11,12,13,14,15);
        lo = *(const v8h*)(arKa + da); hi = *(const v8h*)(arKa + da + 16);
        v16h aKa = __builtin_shufflevector(lo, hi, 0,1,2,3,4,5,6,7,8,9,10,11,12,13,14,15);
        lo = *(const v8h*)(arKb + da); hi = *(const v8h*)(arKb + da + 16);
        v16h aKb = __builtin_shufflevector(lo, hi, 0,1,2,3,4,5,6,7,8,9,10,11,12,13,14,15);
        // B fragments from LDS (shared by all 8 waves): lane K = {o..o+7, o+16..o+23}
        lo = *(const v8h*)(sb0 + d0 + koff8); hi = *(const v8h*)(sb0 + d0 + koff8 + 16);
        v16h b0 = __builtin_shufflevector(lo, hi, 0,1,2,3,4,5,6,7,8,9,10,11,12,13,14,15);
        lo = *(const v8h*)(sb1 + d0 + koff8); hi = *(const v8h*)(sb1 + d0 + koff8 + 16);
        v16h b1 = __builtin_shufflevector(lo, hi, 0,1,2,3,4,5,6,7,8,9,10,11,12,13,14,15);

        cQa0 = __builtin_amdgcn_wmma_f32_16x16x32_f16(false, aQa, false, b0, (short)0, cQa0, false, false);
        cQa1 = __builtin_amdgcn_wmma_f32_16x16x32_f16(false, aQa, false, b1, (short)0, cQa1, false, false);
        cQb0 = __builtin_amdgcn_wmma_f32_16x16x32_f16(false, aQb, false, b0, (short)0, cQb0, false, false);
        cQb1 = __builtin_amdgcn_wmma_f32_16x16x32_f16(false, aQb, false, b1, (short)0, cQb1, false, false);
        cKa0 = __builtin_amdgcn_wmma_f32_16x16x32_f16(false, aKa, false, b0, (short)0, cKa0, false, false);
        cKa1 = __builtin_amdgcn_wmma_f32_16x16x32_f16(false, aKa, false, b1, (short)0, cKa1, false, false);
        cKb0 = __builtin_amdgcn_wmma_f32_16x16x32_f16(false, aKb, false, b0, (short)0, cKb0, false, false);
        cKb1 = __builtin_amdgcn_wmma_f32_16x16x32_f16(false, aKb, false, b1, (short)0, cKb1, false, false);
    }

    // C layout: VGPR r -> M = r (lanes 0-15) or 8+r (lanes 16-31); lane -> N
    const int ebase = (lane < 16) ? 0 : 8;
    const int pc    = lane & 15;
    float* qa0 = q  + ((size_t)(b * DK + e0a + ebase)) * PIX + p0 + pc;
    float* qb0 = q  + ((size_t)(b * DK + e0b + ebase)) * PIX + p0 + pc;
    float* ka0 = kp + ((size_t)(b * DK + e0a + ebase)) * PIX + p0 + pc;
    float* kb0 = kp + ((size_t)(b * DK + e0b + ebase)) * PIX + p0 + pc;
#pragma unroll
    for (int r = 0; r < 8; ++r) {
        size_t ro = (size_t)r * PIX;
        qa0[ro]      = cQa0[r];
        qa0[ro + 16] = cQa1[r];
        qb0[ro]      = cQb0[r];
        qb0[ro + 16] = cQb1[r];
        ka0[ro]      = cKa0[r];
        ka0[ro + 16] = cKa1[r];
        kb0[ro]      = cKb0[r];
        kb0[ro + 16] = cKb1[r];
    }
}

// ------------------------------------------------------------------
// 4) v projection: vp[b,e,p] = sum_d fm[b,d,p]*V[d,e]   (32x32, VALU)
// ------------------------------------------------------------------
__global__ void k_proj_v(const float* __restrict__ fm,
                         const float* __restrict__ V,
                         float* __restrict__ vp) {
    int idx = blockIdx.x * blockDim.x + threadIdx.x;  // ((b*DV+e)*PIX)+p
    if (idx >= BN * DV * PIX) return;
    int p = idx & (PIX - 1);
    int e = (idx >> 12) & (DV - 1);
    int b = idx >> 17;
    const float* src = fm + ((size_t)b * DV) * PIX + p;
    float acc = 0.f;
#pragma unroll
    for (int d = 0; d < DV; ++d)
        acc = fmaf(src[(size_t)d * PIX], V[d * DV + e], acc);
    vp[idx] = acc;
}

// ------------------------------------------------------------------
// 5) scores[b,m,p] = SCALE * sum_d q[b,d,p] * k[m,d,p]
//    k: m<4 -> kp[m], m>=4 -> key_buffer[m-4]   (coalesced over p)
// ------------------------------------------------------------------
__global__ __launch_bounds__(256)
void k_scores(const float* __restrict__ q,
              const float* __restrict__ kp,
              const float* __restrict__ kbuf,
              float* __restrict__ sc, float scale) {
    int b = blockIdx.x >> 4;                       // 0..3
    int p = (blockIdx.x & 15) * 256 + threadIdx.x; // 0..4095
    const float* qs = q + ((size_t)b * DK) * PIX + p;
    for (int m = 0; m < MN; ++m) {
        const float* ks = (m < BN)
            ? kp   + ((size_t)m * DK) * PIX + p
            : kbuf + ((size_t)(m - BN) * DK) * PIX + p;
        // pull next m-slice (key_buffer is the big 117MB HBM stream)
        if (m + 1 < MN && m + 1 >= BN)
            __builtin_prefetch((const void*)(kbuf + ((size_t)(m + 1 - BN) * DK) * PIX + p), 0, 1);
        float acc = 0.f;
#pragma unroll 8
        for (int d = 0; d < DK; ++d)
            acc = fmaf(qs[(size_t)d * PIX], ks[(size_t)d * PIX], acc);
        sc[((size_t)(b * MN + m)) * PIX + p] = scale * acc;
    }
}

// ------------------------------------------------------------------
// 6) softmax over the WIDTH axis (64), in place. One wave per row.
// ------------------------------------------------------------------
__global__ __launch_bounds__(256)
void k_softmax_w(float* __restrict__ sc) {
    int lane = threadIdx.x & 31;
    int wave = threadIdx.x >> 5;
    int row  = blockIdx.x * 8 + wave;              // (b*MN+m)*HS + h, 8192 rows
    float* base = sc + (size_t)row * WS;
    float x0 = base[lane];
    float x1 = base[lane + 32];
    float mx = fmaxf(x0, x1);
#pragma unroll
    for (int off = 16; off > 0; off >>= 1)
        mx = fmaxf(mx, __shfl_xor(mx, off, 32));
    float e0 = __expf(x0 - mx);
    float e1 = __expf(x1 - mx);
    float s = e0 + e1;
#pragma unroll
    for (int off = 16; off > 0; off >>= 1)
        s += __shfl_xor(s, off, 32);
    float inv = 1.0f / s;
    base[lane]      = e0 * inv;
    base[lane + 32] = e1 * inv;
}

// ------------------------------------------------------------------
// 7) out[b,e,p] = fm[b,e,p] + ALPHA * sum_m attn[b,m,p]*v[m,e,p]
// ------------------------------------------------------------------
__global__ __launch_bounds__(256)
void k_ctx_out(const float* __restrict__ attn,
               const float* __restrict__ vp,
               const float* __restrict__ vbuf,
               const float* __restrict__ fm,
               float* __restrict__ out) {
    int idx = blockIdx.x * blockDim.x + threadIdx.x;  // ((b*DV+e)*PIX)+p
    if (idx >= BN * DV * PIX) return;
    int p = idx & (PIX - 1);
    int e = (idx >> 12) & (DV - 1);
    int b = idx >> 17;
    float acc = 0.f;
#pragma unroll 4
    for (int m = 0; m < MN; ++m) {
        float a = attn[((size_t)(b * MN + m)) * PIX + p];
        float vv = (m < BN)
            ? vp  [((size_t)(m * DV + e)) * PIX + p]
            : vbuf[((size_t)((m - BN) * DV + e)) * PIX + p];
        acc = fmaf(a, vv, acc);
    }
    out[idx] = fm[idx] + 0.5f * acc;   // ALPHA = 0.5
}

// ------------------------------------------------------------------
extern "C" void kernel_launch(void* const* d_in, const int* in_sizes, int n_in,
                              void* d_out, int out_size, void* d_ws, size_t ws_size,
                              hipStream_t stream) {
    const float* fc   = (const float*)d_in[0];
    const float* fm   = (const float*)d_in[1];
    const float* kbuf = (const float*)d_in[2];
    const float* vbuf = (const float*)d_in[3];
    const float* Q    = (const float*)d_in[4];
    const float* K    = (const float*)d_in[5];
    const float* V    = (const float*)d_in[6];
    float* out = (float*)d_out;

    char* ws = (char*)d_ws;
    size_t off = 0;
    _Float16* fcT = (_Float16*)(ws + off); off += (size_t)BN * PIX * DK * 2;  // 8.39 MB
    _Float16* Qt  = (_Float16*)(ws + off); off += (size_t)DK * DK * 2;        // 128 KB
    _Float16* Kt  = (_Float16*)(ws + off); off += (size_t)DK * DK * 2;        // 128 KB
    float* q  = (float*)(ws + off); off += (size_t)BN * DK * PIX * 4;         // 16.8 MB
    float* kp = (float*)(ws + off); off += (size_t)BN * DK * PIX * 4;         // 16.8 MB
    float* vp = (float*)(ws + off); off += (size_t)BN * DV * PIX * 4;         // 2.1 MB
    float* sc = (float*)(ws + off); off += (size_t)BN * MN * PIX * 4;         // 2.1 MB

    // SCALE = log(L*h*w + h*w, 1000) / sqrt(DK)
    float scale = (float)(log((double)(32 * HS * WS + HS * WS)) / log(1000.0) / 16.0);

    k_transpose_w <<<dim3((DK * DK) / 256), dim3(256), 0, stream>>>(Q, K, Qt, Kt);
    k_transpose_fc<<<dim3(PIX / 32, DK / 32, BN), dim3(32, 8), 0, stream>>>(fc, fcT);
    k_proj_qk     <<<dim3(BN * (PIX / 32)), dim3(256), 0, stream>>>(fcT, Qt, Kt, q, kp);
    k_proj_v      <<<dim3((BN * DV * PIX) / 256), dim3(256), 0, stream>>>(fm, V, vp);
    k_scores      <<<dim3(BN * (PIX / 256)), dim3(256), 0, stream>>>(q, kp, kbuf, sc, scale);
    k_softmax_w   <<<dim3((BN * MN * HS) / 8), dim3(256), 0, stream>>>(sc);
    k_ctx_out     <<<dim3((BN * DV * PIX) / 256), dim3(256), 0, stream>>>(sc, vp, vbuf, fm, out);
}
